// MultiHeadAttention_59863254172696
// MI455X (gfx1250) — compile-verified, matrix operand order checked by reading, not run on hardware
//
#include <hip/hip_runtime.h>
#include <hip/hip_bf16.h>
#include <cstdint>

// ---------------------------------------------------------------------------
// MI455X (gfx1250) adaLN + MHA pipeline.
//   * all GEMM-like math on v_wmma_f32_16x16x32_f16 (f32 accumulate)
//   * weight tiles staged to LDS via the Tensor Data Mover (tensor_load_to_lds,
//     TENSORcnt, double-buffered); each B fragment feeds 2 WMMAs (2 M-strips)
//   * attention K/V tiles staged via global_load_async_to_lds_b128 (ASYNCcnt)
//   * online softmax in f32 with 16-lane butterflies (wave32)
//   * 1/TEMPERATURE folded into the Q projection epilogue
// Shapes fixed by the reference: B=4, S=2048, D=1024, H=16, Dk=64, T=64.
// TEMPERATURE = 1024 // (16**0.5) = 256.0.
// ---------------------------------------------------------------------------

typedef _Float16     v4h   __attribute__((ext_vector_type(4)));
typedef _Float16     v8h   __attribute__((ext_vector_type(8)));
typedef _Float16     v16h  __attribute__((ext_vector_type(16)));
typedef float        v8f   __attribute__((ext_vector_type(8)));
typedef unsigned int u32x4 __attribute__((ext_vector_type(4)));
typedef int          i32x4 __attribute__((ext_vector_type(4)));
typedef int          i32x8 __attribute__((ext_vector_type(8)));

union AF { v16h v; v8h h[2]; _Float16 e[16]; };

static constexpr int   kD     = 1024;
static constexpr int   kS     = 2048;
static constexpr int   kB     = 4;
static constexpr int   kH     = 16;
static constexpr int   kDk    = 64;
static constexpr int   kM     = kB * kS;       // 8192 rows
static constexpr float kLnEps = 1e-5f;

// ---------------- workspace layout (bytes, all 16B+ aligned) ----------------
static constexpr size_t OFF_SCALE = 0;                                    // 4*1024 f32
static constexpr size_t OFF_SHIFT = OFF_SCALE + (size_t)kB * kD * 4;      // 4*1024 f32
static constexpr size_t OFF_RES   = 32768;                                // 8192*1024 f32
static constexpr size_t OFF_Q16   = OFF_RES  + (size_t)kM * kD * 4;       // f16
static constexpr size_t OFF_WQ16  = OFF_Q16  + (size_t)kM * kD * 2;
static constexpr size_t OFF_WK16  = OFF_WQ16 + (size_t)kD * kD * 2;
static constexpr size_t OFF_WV16  = OFF_WK16 + (size_t)kD * kD * 2;
static constexpr size_t OFF_WF16  = OFF_WV16 + (size_t)kD * kD * 2;
static constexpr size_t OFF_QH    = OFF_WF16 + (size_t)kD * kD * 2;       // [b,h,s,dk] f16
static constexpr size_t OFF_KH    = OFF_QH   + (size_t)kM * kD * 2;       // [b,h,s,dk] f16
static constexpr size_t OFF_VT    = OFF_KH   + (size_t)kM * kD * 2;       // [b,h,dk,s] f16
static constexpr size_t OFF_O16   = OFF_VT   + (size_t)kM * kD * 2;       // [b,s,h*64+dk] f16

// ---------------------------------------------------------------------------
// 1) emb = gelu_exact(t) @ ada_w.T + ada_b ; split into scale / shift
// ---------------------------------------------------------------------------
__global__ void mha_ada_emb(const float* __restrict__ t,
                            const float* __restrict__ ada_w,
                            const float* __restrict__ ada_b,
                            float* __restrict__ scale,
                            float* __restrict__ shift) {
  int gid = blockIdx.x * blockDim.x + threadIdx.x;     // 0..8191
  int b = gid >> 11;
  int j = gid & 2047;
  const float* tb = t + b * 64;
  const float* wr = ada_w + (size_t)j * 64;
  float acc = ada_b[j];
#pragma unroll
  for (int i = 0; i < 64; ++i) {
    float x = tb[i];
    float gl = 0.5f * x * (1.0f + erff(x * 0.70710678118654752f));
    acc = fmaf(gl, wr[i], acc);
  }
  if (j < kD) scale[b * kD + j]        = acc;
  else        shift[b * kD + (j - kD)] = acc;
}

// ---------------------------------------------------------------------------
// 2) adaLN: ln(x)*(1+scale)+shift -> f32 residual + f16 activations
// ---------------------------------------------------------------------------
__global__ void mha_adaln(const float* __restrict__ q,
                          const float* __restrict__ scale,
                          const float* __restrict__ shift,
                          float* __restrict__ resid,
                          _Float16* __restrict__ q16) {
  const int row = blockIdx.x;
  const int b   = row >> 11;
  const float4 xv = ((const float4*)(q + (size_t)row * kD))[threadIdx.x];
  float s  = xv.x + xv.y + xv.z + xv.w;
  float s2 = xv.x * xv.x + xv.y * xv.y + xv.z * xv.z + xv.w * xv.w;
#pragma unroll
  for (int m = 1; m < 32; m <<= 1) {
    s  += __shfl_xor(s,  m, 32);
    s2 += __shfl_xor(s2, m, 32);
  }
  __shared__ float sh1[8], sh2[8];
  const int wave = threadIdx.x >> 5, lane = threadIdx.x & 31;
  if (lane == 0) { sh1[wave] = s; sh2[wave] = s2; }
  __syncthreads();
  if (wave == 0) {
    float a = (lane < 8) ? sh1[lane] : 0.0f;
    float c = (lane < 8) ? sh2[lane] : 0.0f;
#pragma unroll
    for (int m = 1; m < 8; m <<= 1) {
      a += __shfl_xor(a, m, 32);
      c += __shfl_xor(c, m, 32);
    }
    if (lane == 0) { sh1[0] = a; sh2[0] = c; }
  }
  __syncthreads();
  const float mean = sh1[0] * (1.0f / kD);
  const float var  = sh2[0] * (1.0f / kD) - mean * mean;
  const float rstd = rsqrtf(var + kLnEps);

  const int c0 = threadIdx.x * 4;
  const float* sc = scale + b * kD;
  const float* sf = shift + b * kD;
  float vals[4] = {xv.x, xv.y, xv.z, xv.w};
  float4 ro;
  v4h    ho;
#pragma unroll
  for (int i = 0; i < 4; ++i) {
    float ln = (vals[i] - mean) * rstd;
    float y  = ln * (1.0f + sc[c0 + i]) + sf[c0 + i];
    ((float*)&ro)[i] = y;
    ho[i] = (_Float16)y;
  }
  ((float4*)(resid + (size_t)row * kD))[threadIdx.x] = ro;
  ((v4h*)(q16 + (size_t)row * kD))[threadIdx.x]      = ho;
}

// ---------------------------------------------------------------------------
// 3) f32 -> f16 conversion (weights)
// ---------------------------------------------------------------------------
__global__ void mha_cvt_f16(const float* __restrict__ src,
                            _Float16* __restrict__ dst, int n4) {
  int i = blockIdx.x * blockDim.x + threadIdx.x;
  if (i < n4) {
    float4 x = ((const float4*)src)[i];
    v4h h;
    h[0] = (_Float16)x.x; h[1] = (_Float16)x.y;
    h[2] = (_Float16)x.z; h[3] = (_Float16)x.w;
    ((v4h*)dst)[i] = h;
  }
}

// ---------------------------------------------------------------------------
// 4) GEMM  out = (A[8192x1024] @ W[1024x1024]^T) * oscale
//    Block = 8 waves, tile 256(M) x 64(N); each wave owns two 16-row strips,
//    so every LDS B fragment feeds two WMMAs.  The shared 64x64 f16 weight
//    slab is staged by the Tensor Data Mover (double-buffered, wave 0 issues,
//    s_wait_tensorcnt syncs, workgroup barrier publishes).
//    OUT: 0 = f16 head-major [b,h,s,dk]   (Q/K projections; Q gets 1/256)
//         1 = f16 transposed [b,h,dk,s]   (V projection)
//         2 = f32 + residual              (final FC)
// ---------------------------------------------------------------------------
template <bool A_F16, int OUT>
__global__ void __launch_bounds__(256)
mha_gemm(const void* __restrict__ Ap, const _Float16* __restrict__ W,
         void* __restrict__ Out, const float* __restrict__ resid,
         float oscale) {
  __shared__ _Float16 bt[2][64 * 64];            // 16 KB double-buffered W slab
  const int lane   = threadIdx.x & 31;
  const int wave   = threadIdx.x >> 5;
  const int lh     = lane & 15;
  const int g      = lane >> 4;                  // 0 or 1 (half-wave)
  const int m_base = blockIdx.x * 256 + wave * 32;
  const int n_base = blockIdx.y * 64;

  // TDM: 2D tile 64 rows x 64 f16, row stride kD elements, dest contiguous LDS
  auto tdm_issue = [&](int ks, int buf) {
    if (wave == 0) {
      const uint64_t ga = (uint64_t)(uintptr_t)(W + (size_t)n_base * kD + ks);
      const uint32_t la = (uint32_t)(uintptr_t)&bt[buf][0];
      u32x4 g0;
      g0[0] = 1u;                                            // count=1 (valid)
      g0[1] = la;                                            // lds_addr
      g0[2] = (uint32_t)ga;                                  // global_addr lo
      g0[3] = (uint32_t)((ga >> 32) & 0x01FFFFFFu) | (2u << 30); // hi | type=2
      i32x8 g1;
      g1[0] = 0x00010000;           // data_size = 2 bytes
      g1[1] = (int)(1024u << 16);   // tensor_dim0 = 1024 (lo16 in [31:16])
      g1[2] = (int)(1024u << 16);   // tensor_dim0 hi=0 | tensor_dim1 lo = 1024
      g1[3] = (int)(64u << 16);     // tensor_dim1 hi=0 | tile_dim0 = 64
      g1[4] = 64;                   // tile_dim1 = 64, tile_dim2 = 0
      g1[5] = 1024;                 // tensor_dim0_stride = 1024 (lo32)
      g1[6] = 0;                    // stride hi | tensor_dim1_stride lo
      g1[7] = 0;
      i32x4 z4 = {};
#if defined(__clang_major__) && (__clang_major__ >= 23)
      i32x8 z8 = {};
      __builtin_amdgcn_tensor_load_to_lds(g0, g1, z4, z4, z8, 0);
#else
      __builtin_amdgcn_tensor_load_to_lds(g0, g1, z4, z4, 0);
#endif
    }
  };

  v8f c[2][4] = {};
  const _Float16* A16 = (const _Float16*)Ap;
  const float*    A32 = (const float*)Ap;

  tdm_issue(0, 0);
  for (int ks = 0; ks < kD; ks += 64) {
    const int buf = (ks >> 6) & 1;
    if (ks + 64 < kD) {
      tdm_issue(ks + 64, buf ^ 1);                       // prefetch next slab
      if (wave == 0) __builtin_amdgcn_s_wait_tensorcnt(1);
      if constexpr (A_F16)
        __builtin_prefetch(A16 + (size_t)(m_base + lh) * kD + ks + 64, 0, 1);
      else
        __builtin_prefetch(A32 + (size_t)(m_base + lh) * kD + ks + 64, 0, 1);
    } else {
      if (wave == 0) __builtin_amdgcn_s_wait_tensorcnt(0);
    }
    __syncthreads();                                     // slab visible to all

#pragma unroll
    for (int half = 0; half < 2; ++half) {
      const int k = ks + half * 32;
      AF a[2];
#pragma unroll
      for (int ms = 0; ms < 2; ++ms) {
        if constexpr (A_F16) {
          const _Float16* ap =
              A16 + (size_t)(m_base + ms * 16 + lh) * kD + k + g * 8;
          a[ms].h[0] = *(const v8h*)ap;
          a[ms].h[1] = *(const v8h*)(ap + 16);
        } else {
          const float* ap =
              A32 + (size_t)(m_base + ms * 16 + lh) * kD + k + g * 8;
#pragma unroll
          for (int i = 0; i < 8; ++i) {
            a[ms].e[i]     = (_Float16)ap[i];
            a[ms].e[8 + i] = (_Float16)ap[i + 16];
          }
        }
      }
#pragma unroll
      for (int nt = 0; nt < 4; ++nt) {
        AF bf;
        const _Float16* bp = &bt[buf][(nt * 16 + lh) * 64 + half * 32 + g * 8];
        bf.h[0] = *(const v8h*)bp;
        bf.h[1] = *(const v8h*)(bp + 16);
#pragma unroll
        for (int ms = 0; ms < 2; ++ms) {
          c[ms][nt] = __builtin_amdgcn_wmma_f32_16x16x32_f16(
              false, a[ms].v, false, bf.v, (short)0, c[ms][nt], false, false);
        }
      }
    }
    __syncthreads();                  // protect slab from next TDM overwrite
  }

  if constexpr (OUT == 0) {            // [b,h,s,dk] f16, scaled
    _Float16* o = (_Float16*)Out;
#pragma unroll
    for (int ms = 0; ms < 2; ++ms) {
#pragma unroll
      for (int nt = 0; nt < 4; ++nt) {
        const int j = n_base + nt * 16 + lh;
        const int h = j >> 6, dk = j & 63;
#pragma unroll
        for (int r = 0; r < 8; ++r) {
          const int m  = m_base + ms * 16 + g * 8 + r;
          const int bb = m >> 11, s = m & 2047;
          o[(((size_t)(bb * kH + h)) * kS + s) * kDk + dk] =
              (_Float16)(c[ms][nt][r] * oscale);
        }
      }
    }
  } else if constexpr (OUT == 1) {     // [b,h,dk,s] f16 (packed b128 store)
    _Float16* o = (_Float16*)Out;
#pragma unroll
    for (int ms = 0; ms < 2; ++ms) {
      const int mb = m_base + ms * 16;
      const int bb = mb >> 11;
      const int s0 = (mb & 2047) + g * 8;
#pragma unroll
      for (int nt = 0; nt < 4; ++nt) {
        const int j = n_base + nt * 16 + lh;
        const int h = j >> 6, dk = j & 63;
        v8h pk;
#pragma unroll
        for (int r = 0; r < 8; ++r) pk[r] = (_Float16)(c[ms][nt][r] * oscale);
        *(v8h*)(o + (((size_t)(bb * kH + h)) * kDk + dk) * kS + s0) = pk;
      }
    }
  } else {                             // f32 + residual
    float* o = (float*)Out;
#pragma unroll
    for (int ms = 0; ms < 2; ++ms) {
#pragma unroll
      for (int nt = 0; nt < 4; ++nt) {
        const int j = n_base + nt * 16 + lh;
#pragma unroll
        for (int r = 0; r < 8; ++r) {
          const size_t idx = (size_t)(m_base + ms * 16 + g * 8 + r) * kD + j;
          o[idx] = c[ms][nt][r] + resid[idx];
        }
      }
    }
  }
}

// ---------------------------------------------------------------------------
// 5) Flash attention, 64-key chunks.  All 8 waves of a block share the same
//    K/V chunk, staged once per block into LDS with async loads
//    (global_load_async_to_lds_b128, ASYNCcnt, double buffered).  Per chunk:
//    8 WMMA for QK^T (Q pre-scaled by 1/256), online softmax (16-lane
//    butterflies, exp in f32), P bounced through a wave-private LDS tile,
//    8 WMMA for PV.  Block = 8 waves -> 128 Q rows; grid = (B*H, S/128).
// ---------------------------------------------------------------------------
__global__ void __launch_bounds__(256)
mha_attn(const _Float16* __restrict__ qh, const _Float16* __restrict__ kh,
         const _Float16* __restrict__ vt, _Float16* __restrict__ o16) {
  __shared__ _Float16 ktile[2][64 * 64];   // [key][d]   8 KB x2
  __shared__ _Float16 vtile[2][64 * 64];   // [d][key]   8 KB x2
  __shared__ _Float16 plds[8][16 * 64];    // per-wave P tile, 16 KB
  const int tid  = threadIdx.x;
  const int lane = tid & 31, wave = tid >> 5;
  const int lh = lane & 15, g = lane >> 4;
  const int bh = blockIdx.x;                       // 0..63
  const int qb = blockIdx.y * 128 + wave * 16;
  const _Float16* kbase = kh + (size_t)bh * kS * kDk;
  const _Float16* vbase = vt + (size_t)bh * kDk * kS;

  const int trow = tid >> 2;                       // 0..63
  const int tcol = (tid & 3) * 16;                 // 0,16,32,48

  // each wave async-copies its 1/8 slice of the shared K and Vt chunk tiles
  auto stage = [&](int k0, int buf) {
    uint32_t gk = (uint32_t)(((k0 + trow) * kDk + tcol) * 2);
    uint32_t lk = (uint32_t)(uintptr_t)&ktile[buf][trow * 64 + tcol];
    asm volatile("global_load_async_to_lds_b128 %0, %1, %2"
                 :: "v"(lk), "v"(gk), "s"(kbase) : "memory");
    asm volatile("global_load_async_to_lds_b128 %0, %1, %2"
                 :: "v"(lk + 16u), "v"(gk + 16u), "s"(kbase) : "memory");
    uint32_t gv = (uint32_t)((trow * kS + k0 + tcol) * 2);
    uint32_t lv = (uint32_t)(uintptr_t)&vtile[buf][trow * 64 + tcol];
    asm volatile("global_load_async_to_lds_b128 %0, %1, %2"
                 :: "v"(lv), "v"(gv), "s"(vbase) : "memory");
    asm volatile("global_load_async_to_lds_b128 %0, %1, %2"
                 :: "v"(lv + 16u), "v"(gv + 16u), "s"(vbase) : "memory");
  };

  // Q fragments for the whole strip (Dk=64 -> two 16x32 A fragments)
  AF aq[2];
#pragma unroll
  for (int kt = 0; kt < 2; ++kt) {
    const _Float16* ap = qh + (size_t)bh * kS * kDk +
                         (size_t)(qb + lh) * kDk + kt * 32 + g * 8;
    aq[kt].h[0] = *(const v8h*)ap;
    aq[kt].h[1] = *(const v8h*)(ap + 16);
  }

  v8f acc[4] = {};
  float run_m[8], run_s[8];
#pragma unroll
  for (int r = 0; r < 8; ++r) { run_m[r] = -1e30f; run_s[r] = 0.0f; }

  stage(0, 0);
  for (int cch = 0; cch < kS / 64; ++cch) {        // 32 chunks of 64 keys
    const int buf = cch & 1;
    if (cch + 1 < kS / 64) {
      stage((cch + 1) * 64, buf ^ 1);              // prefetch next chunk
      asm volatile("s_wait_asynccnt 0x4" ::: "memory");  // older 4 done
    } else {
      asm volatile("s_wait_asynccnt 0x0" ::: "memory");
    }
    __syncthreads();

    // ---- scores: 4 key tiles x 2 d-substeps = 8 WMMA ----
    v8f sc[4] = {};
#pragma unroll
    for (int nt = 0; nt < 4; ++nt) {
#pragma unroll
      for (int kt = 0; kt < 2; ++kt) {
        AF bk;
        const _Float16* bp = &ktile[buf][(nt * 16 + lh) * 64 + kt * 32 + g * 8];
        bk.h[0] = *(const v8h*)bp;
        bk.h[1] = *(const v8h*)(bp + 16);
        sc[nt] = __builtin_amdgcn_wmma_f32_16x16x32_f16(
            false, aq[kt].v, false, bk.v, (short)0, sc[nt], false, false);
      }
    }

    // ---- online softmax over this 64-key chunk (scores pre-scaled) ----
    float corr[8];
#pragma unroll
    for (int r = 0; r < 8; ++r) {
      float mx = fmaxf(fmaxf(sc[0][r], sc[1][r]), fmaxf(sc[2][r], sc[3][r]));
#pragma unroll
      for (int msk = 1; msk < 16; msk <<= 1)
        mx = fmaxf(mx, __shfl_xor(mx, msk, 32));
      const float nm = fmaxf(run_m[r], mx);
      corr[r]  = __expf(run_m[r] - nm);
      run_m[r] = nm;
    }
    float rs[8];
#pragma unroll
    for (int r = 0; r < 8; ++r) rs[r] = 0.0f;
#pragma unroll
    for (int nt = 0; nt < 4; ++nt) {
#pragma unroll
      for (int r = 0; r < 8; ++r) {
        const float p = __expf(sc[nt][r] - run_m[r]);
        rs[r] += p;
        plds[wave][(g * 8 + r) * 64 + nt * 16 + lh] = (_Float16)p;
      }
    }
#pragma unroll
    for (int r = 0; r < 8; ++r) {
      float s = rs[r];
#pragma unroll
      for (int msk = 1; msk < 16; msk <<= 1) s += __shfl_xor(s, msk, 32);
      run_s[r] = run_s[r] * corr[r] + s;
#pragma unroll
      for (int dt = 0; dt < 4; ++dt) acc[dt][r] *= corr[r];
    }

    // re-read P as two 16x32 WMMA A fragments (K = key index)
    AF pa[2];
#pragma unroll
    for (int kt = 0; kt < 2; ++kt) {
      const _Float16* lp = &plds[wave][lh * 64 + kt * 32 + g * 8];
      pa[kt].h[0] = *(const v8h*)lp;
      pa[kt].h[1] = *(const v8h*)(lp + 16);
    }

    // ---- PV: 4 d-tiles x 2 key-substeps = 8 WMMA ----
#pragma unroll
    for (int dt = 0; dt < 4; ++dt) {
#pragma unroll
      for (int kt = 0; kt < 2; ++kt) {
        AF bv;
        const _Float16* bp = &vtile[buf][(dt * 16 + lh) * 64 + kt * 32 + g * 8];
        bv.h[0] = *(const v8h*)bp;
        bv.h[1] = *(const v8h*)(bp + 16);
        acc[dt] = __builtin_amdgcn_wmma_f32_16x16x32_f16(
            false, pa[kt].v, false, bv.v, (short)0, acc[dt], false, false);
      }
    }
    __syncthreads();                  // protect tiles from next prefetch
  }

  // ---- epilogue: normalize, store f16 in [b, s, h*64+dk] layout ----
  const int bb = bh >> 4, h = bh & 15;
#pragma unroll
  for (int r = 0; r < 8; ++r) {
    const float inv  = 1.0f / run_s[r];
    const int   qrow = qb + g * 8 + r;
    _Float16* op = o16 + ((size_t)(bb * kS + qrow)) * kD + h * kDk;
#pragma unroll
    for (int dt = 0; dt < 4; ++dt)
      op[dt * 16 + lh] = (_Float16)(acc[dt][r] * inv);
  }
}

// ---------------------------------------------------------------------------
extern "C" void kernel_launch(void* const* d_in, const int* in_sizes, int n_in,
                              void* d_out, int out_size, void* d_ws, size_t ws_size,
                              hipStream_t stream) {
  const float* q     = (const float*)d_in[0];
  const float* k     = (const float*)d_in[1];
  const float* v     = (const float*)d_in[2];
  const float* t     = (const float*)d_in[3];
  const float* w_q   = (const float*)d_in[4];
  const float* w_k   = (const float*)d_in[5];
  const float* w_v   = (const float*)d_in[6];
  const float* w_fc  = (const float*)d_in[7];
  const float* ada_w = (const float*)d_in[8];
  const float* ada_b = (const float*)d_in[9];
  float* out = (float*)d_out;

  char* ws = (char*)d_ws;
  float*     scale = (float*)(ws + OFF_SCALE);
  float*     shift = (float*)(ws + OFF_SHIFT);
  float*     resid = (float*)(ws + OFF_RES);
  _Float16*  q16   = (_Float16*)(ws + OFF_Q16);
  _Float16*  wq16  = (_Float16*)(ws + OFF_WQ16);
  _Float16*  wk16  = (_Float16*)(ws + OFF_WK16);
  _Float16*  wv16  = (_Float16*)(ws + OFF_WV16);
  _Float16*  wf16  = (_Float16*)(ws + OFF_WF16);
  _Float16*  qh    = (_Float16*)(ws + OFF_QH);
  _Float16*  kh    = (_Float16*)(ws + OFF_KH);
  _Float16*  vt    = (_Float16*)(ws + OFF_VT);
  _Float16*  o16   = (_Float16*)(ws + OFF_O16);

  // 1) adaLN modulation vectors
  mha_ada_emb<<<(kB * 2 * kD) / 256, 256, 0, stream>>>(t, ada_w, ada_b, scale, shift);
  // 2) adaLN over q -> residual (f32) + activations (f16)
  mha_adaln<<<kM, 256, 0, stream>>>(q, scale, shift, resid, q16);
  // 3) weights -> f16
  const int w4 = (kD * kD) / 4;
  mha_cvt_f16<<<(w4 + 255) / 256, 256, 0, stream>>>(w_q,  wq16, w4);
  mha_cvt_f16<<<(w4 + 255) / 256, 256, 0, stream>>>(w_k,  wk16, w4);
  mha_cvt_f16<<<(w4 + 255) / 256, 256, 0, stream>>>(w_v,  wv16, w4);
  mha_cvt_f16<<<(w4 + 255) / 256, 256, 0, stream>>>(w_fc, wf16, w4);

  // 4) projections (WMMA + TDM weight staging); Q absorbs 1/TEMPERATURE
  dim3 gg(kM / 256, kD / 64);   // (32, 16)
  mha_gemm<true,  0><<<gg, 256, 0, stream>>>(q16, wq16, qh, nullptr, 1.0f / 256.0f);
  mha_gemm<false, 0><<<gg, 256, 0, stream>>>(k,   wk16, kh, nullptr, 1.0f);
  mha_gemm<false, 1><<<gg, 256, 0, stream>>>(v,   wv16, vt, nullptr, 1.0f);

  // 5) flash attention (WMMA + async K/V staging)
  mha_attn<<<dim3(kB * kH, kS / 128), 256, 0, stream>>>(qh, kh, vt, o16);

  // 6) FC + residual -> f32 output (WMMA + TDM weight staging)
  mha_gemm<true, 2><<<gg, 256, 0, stream>>>(o16, wf16, out, resid, 1.0f);

  (void)in_sizes; (void)n_in; (void)out_size; (void)ws_size;
}
